// QValueNet_38268158607919
// MI455X (gfx1250) — compile-verified, matrix operand order checked by reading, not run on hardware
//
#include <hip/hip_runtime.h>
#include <stdint.h>

#define LRELU_SLOPE 0.01f
#define BN_EPS 1e-5f
#define NE 100000
#define NEP 100032            // NE padded to multiple of 64
#define NNODES 20000

typedef __attribute__((ext_vector_type(16))) __bf16 v16bf;
typedef __attribute__((ext_vector_type(8)))  float  v8f;
typedef __attribute__((ext_vector_type(4)))  unsigned int u32x4;

union Frag { v16bf v; u32x4 u[2]; };

static __device__ __forceinline__ unsigned short f2bf(float f) {
  union { float f; unsigned u; } x; x.f = f;
  unsigned r = x.u + 0x7FFFu + ((x.u >> 16) & 1u);
  return (unsigned short)(r >> 16);
}
static __device__ __forceinline__ float lrelu(float v) {
  return v > 0.f ? v : LRELU_SLOPE * v;
}

// ---------------- conv 3x3 (1->32) + lrelu + segment-sum scatter ----------------
__global__ void k_conv_scatter(const float* __restrict__ raw,
                               const int* __restrict__ seg,
                               const float* __restrict__ cw,
                               const float* __restrict__ cb,
                               float* __restrict__ ssum,
                               float* __restrict__ cnt) {
  int gid = blockIdx.x * blockDim.x + threadIdx.x;
  const int HW = 256 * 256;
  if (gid >= 4 * HW) return;
  int b = gid / HW, rem = gid % HW, y = rem >> 8, x = rem & 255;
  float t[9];
  #pragma unroll
  for (int ky = 0; ky < 3; ky++)
    #pragma unroll
    for (int kx = 0; kx < 3; kx++) {
      int yy = y + ky - 1, xx = x + kx - 1;
      t[ky*3+kx] = (yy >= 0 && yy < 256 && xx >= 0 && xx < 256)
                   ? raw[b*HW + yy*256 + xx] : 0.f;
    }
  int id = seg[gid] + b * 5000;
  for (int c = 0; c < 32; c++) {
    float a = cb[c];
    #pragma unroll
    for (int j = 0; j < 9; j++) a += cw[c*9+j] * t[j];
    atomicAdd(&ssum[id*32 + c], lrelu(a));
  }
  atomicAdd(&cnt[id], 1.0f);
}

// ---------------- node features: [mean_emb | sp_feat] -> bf16 ----------------
__global__ void k_node_feats(const float* __restrict__ ssum, const float* __restrict__ cnt,
                             const float* __restrict__ spf, unsigned short* __restrict__ xbf) {
  int i = blockIdx.x * blockDim.x + threadIdx.x;
  if (i >= NNODES * 64) return;
  int n = i >> 6, c = i & 63;
  float v = (c < 32) ? ssum[n*32 + c] / fmaxf(cnt[n], 1.0f) : spf[n*32 + (c - 32)];
  xbf[i] = f2bf(v);
}

// ---------------- weight f32 [K,N] -> bf16 transposed [N,Kpad] (zero pad) ------
__global__ void k_conv_weight(const float* __restrict__ W, unsigned short* __restrict__ Wt,
                              int K, int N, int Kpad) {
  int i = blockIdx.x * blockDim.x + threadIdx.x;
  if (i >= N * Kpad) return;
  int n = i / Kpad, k = i % Kpad;
  Wt[i] = (k < K) ? f2bf(W[k*N + n]) : (unsigned short)0;
}

// ---------------- assemble message matrix M[NEP,160] bf16 (padded rows = 0) ----
__global__ void k_build_m(const unsigned short* __restrict__ xbf, const int* __restrict__ ei,
                          const float* __restrict__ actions, const float* __restrict__ ef3,
                          unsigned short* __restrict__ Mb) {
  int e = blockIdx.x * blockDim.x + threadIdx.x;
  if (e >= NEP) return;
  unsigned* row = (unsigned*)(Mb + (size_t)e * 160);
  if (e >= NE) {
    #pragma unroll
    for (int i = 0; i < 80; i++) row[i] = 0u;
    return;
  }
  int src = ei[e], dst = ei[NE + e];
  const unsigned* xs = (const unsigned*)(xbf + src*64);
  const unsigned* xd = (const unsigned*)(xbf + dst*64);
  #pragma unroll
  for (int i = 0; i < 32; i++) row[i] = xs[i];
  #pragma unroll
  for (int i = 0; i < 32; i++) row[32 + i] = xd[i];
  row[64] = (unsigned)f2bf(actions[e]) | ((unsigned)f2bf(ef3[e*3+0]) << 16);
  row[65] = (unsigned)f2bf(ef3[e*3+1]) | ((unsigned)f2bf(ef3[e*3+2]) << 16);
  #pragma unroll
  for (int i = 66; i < 80; i++) row[i] = 0u;
}

// =====================================================================================
// LDS-staged bf16 WMMA GEMM: C[Mp,N] = A[Mp,K] x Wt[N,K]^T + bias
//   block = 256 threads = 8 waves (2 x 4), super-tile 64(M) x 256(N)
//   per k-step: stage A 64x32 (4KB) + B 256x32 (16KB) into LDS (stride 40 elems,
//   80B rows -> conflict-free ds_load_b128 fragment reads), 8 WMMAs per wave.
//   Requires: Mp % 64 == 0, N % 256 == 0, K % 32 == 0.  No guards anywhere.
// =====================================================================================
#define LSTR 40
__global__ void __launch_bounds__(256) k_gemm_lds(
    const unsigned short* __restrict__ A, const unsigned short* __restrict__ Bt,
    const float* __restrict__ bias, float* __restrict__ Cf,
    unsigned short* __restrict__ Cbf, int Mp, int N, int K, int act) {
  __shared__ unsigned short As[64 * LSTR];
  __shared__ unsigned short Bs[256 * LSTR];
  int nblocks = N >> 8;
  int mblk = blockIdx.x / nblocks, nblk = blockIdx.x % nblocks;
  int tid = threadIdx.x;
  int wave = tid >> 5, lane = tid & 31;
  int tml = wave >> 2, tnl = wave & 3;          // 2 x 4 wave grid
  int lm = lane & 15, half = lane >> 4, kb = half * 8;

  // staging assignments
  int arow = tid >> 2, achk = (tid & 3) * 8;    // A: 64 rows x 4 b128 chunks
  const unsigned short* gA = A + (size_t)(mblk*64 + arow) * K + achk;
  const unsigned short* gB = Bt + (size_t)(nblk*256 + tid) * K;   // B: 1 col/thread

  v8f acc[2][4] = {};

  for (int k0 = 0; k0 < K; k0 += 32) {
    // ---- global -> LDS stage ----
    *(u32x4*)(As + arow*LSTR + achk) = *(const u32x4*)(gA + k0);
    #pragma unroll
    for (int c = 0; c < 4; c++)
      *(u32x4*)(Bs + tid*LSTR + c*8) = *(const u32x4*)(gB + k0 + c*8);
    if (k0 + 32 < K) {                 // prefetch next k-tile -> global_prefetch_b8
      __builtin_prefetch(gA + k0 + 32, 0, 3);
      __builtin_prefetch(gB + k0 + 32, 0, 3);
    }
    __syncthreads();

    // ---- fragments from LDS + WMMA ----
    Frag a[2];
    #pragma unroll
    for (int t = 0; t < 2; t++) {
      const unsigned short* p = As + (tml*32 + t*16 + lm) * LSTR + kb;
      a[t].u[0] = *(const u32x4*)p;
      a[t].u[1] = *(const u32x4*)(p + 16);
    }
    #pragma unroll
    for (int j = 0; j < 4; j++) {
      const unsigned short* q = Bs + (tnl*64 + j*16 + lm) * LSTR + kb;
      Frag b; b.u[0] = *(const u32x4*)q; b.u[1] = *(const u32x4*)(q + 16);
      #pragma unroll
      for (int t = 0; t < 2; t++)
        acc[t][j] = __builtin_amdgcn_wmma_f32_16x16x32_bf16(
            false, a[t].v, false, b.v, (short)0, acc[t][j], false, false);
    }
    __syncthreads();
  }

  // ---- epilogue (unguarded: Mp padded) ----
  #pragma unroll
  for (int j = 0; j < 4; j++) {
    int n = nblk*256 + tnl*64 + j*16 + lm;
    float bv = bias[n];
    #pragma unroll
    for (int t = 0; t < 2; t++) {
      #pragma unroll
      for (int r = 0; r < 8; r++) {
        int m = mblk*64 + tml*32 + t*16 + r + 8*half;   // C/D: lanes16-31 -> M=8+r
        float v = acc[t][j][r] + bv;
        if (act) v = lrelu(v);
        size_t idx = (size_t)m * N + n;
        if (Cf)  Cf[idx] = v;
        if (Cbf) Cbf[idx] = f2bf(v);
      }
    }
  }
}

// ---------------- simple per-wave WMMA GEMM (used for N=64 layer) ----------------
__global__ void __launch_bounds__(256) k_gemm_bf16(
    const unsigned short* __restrict__ A, const unsigned short* __restrict__ Bt,
    const float* __restrict__ bias, float* __restrict__ Cf,
    unsigned short* __restrict__ Cbf, int M, int N, int K, int act) {
  int wid = blockIdx.x * 8 + (threadIdx.x >> 5);
  int ntiles = N >> 6;
  int mtiles = M >> 5;                       // M % 32 == 0 (padded)
  if (wid >= mtiles * ntiles) return;
  int tm = wid / ntiles, tn = wid % ntiles;
  int lane = threadIdx.x & 31;
  int lm = lane & 15, half = lane >> 4, kb = half * 8;

  v8f acc[2][4] = {};
  for (int k0 = 0; k0 < K; k0 += 32) {
    Frag a[2];
    #pragma unroll
    for (int t = 0; t < 2; t++) {
      const unsigned short* p = A + (size_t)(tm*32 + t*16 + lm) * K + k0 + kb;
      a[t].u[0] = *(const u32x4*)p;
      a[t].u[1] = *(const u32x4*)(p + 16);
    }
    #pragma unroll
    for (int j = 0; j < 4; j++) {
      const unsigned short* q = Bt + (size_t)(tn*64 + j*16 + lm) * K + k0 + kb;
      Frag b; b.u[0] = *(const u32x4*)q; b.u[1] = *(const u32x4*)(q + 16);
      #pragma unroll
      for (int t = 0; t < 2; t++)
        acc[t][j] = __builtin_amdgcn_wmma_f32_16x16x32_bf16(
            false, a[t].v, false, b.v, (short)0, acc[t][j], false, false);
    }
  }
  #pragma unroll
  for (int j = 0; j < 4; j++) {
    int n = tn*64 + j*16 + lm;
    float bv = bias[n];
    #pragma unroll
    for (int t = 0; t < 2; t++) {
      #pragma unroll
      for (int r = 0; r < 8; r++) {
        int m = tm*32 + t*16 + r + 8*half;
        float v = acc[t][j][r] + bv;
        if (act) v = lrelu(v);
        size_t idx = (size_t)m * N + n;
        if (Cf)  Cf[idx] = v;
        if (Cbf) Cbf[idx] = f2bf(v);
      }
    }
  }
}

// ---------------- sigmoid score + MSE side loss (deterministic 2-stage) --------
__global__ void k_score(const float* __restrict__ EF, const float* __restrict__ sw,
                        const float* __restrict__ sb, const float* __restrict__ gt,
                        float* __restrict__ partial) {
  __shared__ float sm[256];
  int e = blockIdx.x * 256 + threadIdx.x;
  float d2 = 0.f;
  if (e < NE) {
    float z = sb[0];
    #pragma unroll
    for (int c = 0; c < 64; c++) z += EF[(size_t)e*64 + c] * sw[c];
    float s = 1.f / (1.f + __expf(-z));
    float d = s - gt[e];
    d2 = d * d;
  }
  sm[threadIdx.x] = d2;
  __syncthreads();
  for (int off = 128; off > 0; off >>= 1) {
    if (threadIdx.x < off) sm[threadIdx.x] += sm[threadIdx.x + off];
    __syncthreads();
  }
  if (threadIdx.x == 0) partial[blockIdx.x] = sm[0];
}
__global__ void k_loss_final(const float* __restrict__ partial, int nb, float* __restrict__ out) {
  if (threadIdx.x == 0 && blockIdx.x == 0) {
    float s = 0.f;
    for (int i = 0; i < nb; i++) s += partial[i];
    out[0] = s * (1.0f / (float)NE);
  }
}

// ---------------- subgraph gather (padded rows -> 0) ----------------
__global__ void k_gather(const float* __restrict__ EF, const int* __restrict__ sub,
                         float* __restrict__ out, int nsub, int npad, int ssg) {
  int d = ssg * 64;
  int i = blockIdx.x * blockDim.x + threadIdx.x;
  if (i >= npad * d) return;
  int r = i / d, c = i % d;
  if (r >= nsub) { out[i] = 0.f; return; }
  int e = sub[r * ssg + (c >> 6)];
  out[i] = EF[(size_t)e * 64 + (c & 63)];
}

// ---------------- batch-norm column stats (population var, deterministic) ------
__global__ void k_bn_stats(const float* __restrict__ X, int n, int d,
                           float* __restrict__ meanb, float* __restrict__ rstdb) {
  int c = blockIdx.x;
  __shared__ float s1[256], s2[256];
  float a = 0.f, b = 0.f;
  for (int r = threadIdx.x; r < n; r += 256) {
    float v = X[(size_t)r * d + c];
    a += v; b += v * v;
  }
  s1[threadIdx.x] = a; s2[threadIdx.x] = b;
  __syncthreads();
  for (int off = 128; off > 0; off >>= 1) {
    if (threadIdx.x < off) { s1[threadIdx.x] += s1[threadIdx.x+off]; s2[threadIdx.x] += s2[threadIdx.x+off]; }
    __syncthreads();
  }
  if (threadIdx.x == 0) {
    float mu = s1[0] / (float)n;
    float var = s2[0] / (float)n - mu * mu;
    meanb[c] = mu;
    rstdb[c] = rsqrtf(var + BN_EPS);
  }
}

// ---------------- bn normalize + lrelu -> bf16 and/or f32 (over padded rows) ---
__global__ void k_bn_apply(const float* __restrict__ X, const float* __restrict__ meanb,
                           const float* __restrict__ rstdb, const float* __restrict__ gamma,
                           const float* __restrict__ beta, unsigned short* __restrict__ outbf,
                           float* __restrict__ outf, int npad, int d) {
  int i = blockIdx.x * blockDim.x + threadIdx.x;
  if (i >= npad * d) return;
  int c = i % d;
  float v = (X[i] - meanb[c]) * rstdb[c] * gamma[c] + beta[c];
  v = lrelu(v);
  if (outbf) outbf[i] = f2bf(v);
  if (outf)  outf[i]  = v;
}

// ---------------- final 512 -> 1 layer: wave-per-row dot ----------------
__global__ void k_matvec512(const float* __restrict__ X, const float* __restrict__ w,
                            const float* __restrict__ b, float* __restrict__ out, int n) {
  int wid = blockIdx.x * 8 + (threadIdx.x >> 5);
  if (wid >= n) return;
  int lane = threadIdx.x & 31;
  float s = 0.f;
  for (int c = lane; c < 512; c += 32) s += X[(size_t)wid * 512 + c] * w[c];
  for (int off = 16; off > 0; off >>= 1) s += __shfl_xor(s, off, 32);
  if (lane == 0) out[wid] = s + b[0];
}

// =================================================================================
extern "C" void kernel_launch(void* const* d_in, const int* in_sizes, int n_in,
                              void* d_out, int out_size, void* d_ws, size_t ws_size,
                              hipStream_t stream) {
  (void)n_in; (void)out_size; (void)ws_size;
  char* ws = (char*)d_ws;
  float* out = (float*)d_out;

  // ---- fixed top-level inputs ----
  const float* raw      = (const float*)d_in[0];
  const int*   sp_seg   = (const int*)d_in[1];
  const float* sp_feat  = (const float*)d_in[2];
  const float* actions  = (const float*)d_in[3];
  const int*   edge_idx = (const int*)d_in[4];
  const float* edge_ft  = (const float*)d_in[5];
  const int*   subg[3]  = { (const int*)d_in[6], (const int*)d_in[7], (const int*)d_in[8] };
  const float* gt       = (const float*)d_in[9];

  // ---- params: detect insertion vs jax-sorted flattening from in_sizes ----
  bool ins = (in_sizes[10] == 288);   // conv_w(288) first => insertion order
  const float *cw, *cb, *g1w, *g1b, *g2w, *g2b, *g3w, *g3b, *scw, *scb;
  const float *bn0g[3], *bn0b[3], *bn1g[3], *bn1b[3], *bn2g[3], *bn2b[3];
  const float *l1w[3], *l1b[3], *l2w[3], *l2b[3], *l3w[3], *l3b[3];
  if (ins) {
    cw=(const float*)d_in[10]; cb=(const float*)d_in[11];
    g1w=(const float*)d_in[12]; g1b=(const float*)d_in[13];
    g2w=(const float*)d_in[14]; g2b=(const float*)d_in[15];
    g3w=(const float*)d_in[16]; g3b=(const float*)d_in[17];
    scw=(const float*)d_in[18]; scb=(const float*)d_in[19];
    for (int i = 0; i < 3; i++) {
      int h = 20 + 12*i;
      bn0g[i]=(const float*)d_in[h+0]; bn0b[i]=(const float*)d_in[h+1];
      l1w[i]=(const float*)d_in[h+2];  l1b[i]=(const float*)d_in[h+3];
      bn1g[i]=(const float*)d_in[h+4]; bn1b[i]=(const float*)d_in[h+5];
      l2w[i]=(const float*)d_in[h+6];  l2b[i]=(const float*)d_in[h+7];
      bn2g[i]=(const float*)d_in[h+8]; bn2b[i]=(const float*)d_in[h+9];
      l3w[i]=(const float*)d_in[h+10]; l3b[i]=(const float*)d_in[h+11];
    }
  } else {            // jax tree-sorted: conv_b, conv_w, g1, g2, g3, heads, score
    cb=(const float*)d_in[10]; cw=(const float*)d_in[11];
    g1w=(const float*)d_in[12]; g1b=(const float*)d_in[13];
    g2w=(const float*)d_in[14]; g2b=(const float*)d_in[15];
    g3w=(const float*)d_in[16]; g3b=(const float*)d_in[17];
    for (int i = 0; i < 3; i++) {
      int h = 18 + 12*i;  // head keys sorted: bn0,bn1,bn2,l1,l2,l3
      bn0g[i]=(const float*)d_in[h+0]; bn0b[i]=(const float*)d_in[h+1];
      bn1g[i]=(const float*)d_in[h+2]; bn1b[i]=(const float*)d_in[h+3];
      bn2g[i]=(const float*)d_in[h+4]; bn2b[i]=(const float*)d_in[h+5];
      l1w[i]=(const float*)d_in[h+6];  l1b[i]=(const float*)d_in[h+7];
      l2w[i]=(const float*)d_in[h+8];  l2b[i]=(const float*)d_in[h+9];
      l3w[i]=(const float*)d_in[h+10]; l3b[i]=(const float*)d_in[h+11];
    }
    scw=(const float*)d_in[54]; scb=(const float*)d_in[55];
  }

  // ---- workspace layout ----
  size_t cur = 0;
  auto alloc = [&](size_t bytes) { size_t o = cur; cur = (cur + bytes + 255) & ~(size_t)255; return o; };
  size_t o_ssum = alloc((size_t)NNODES*32*4);
  size_t o_cnt  = alloc((size_t)NNODES*4);
  size_t o_xbf  = alloc((size_t)NNODES*64*2);
  size_t o_wg1  = alloc((size_t)512*160*2);
  size_t o_wg2  = alloc((size_t)512*512*2);
  size_t o_wg3  = alloc((size_t)64*512*2);
  size_t o_wl1[3], o_wl2[3];
  int dim_d[3] = {256, 512, 1024};
  for (int i = 0; i < 3; i++) o_wl1[i] = alloc((size_t)512*dim_d[i]*2);
  for (int i = 0; i < 3; i++) o_wl2[i] = alloc((size_t)512*512*2);
  size_t o_mean = alloc(1024*4);
  size_t o_rstd = alloc(1024*4);
  size_t o_part = alloc(512*4);
  size_t o_m    = alloc((size_t)NEP*160*2);
  size_t o_h1   = alloc((size_t)NEP*512*2);
  size_t o_h2   = alloc((size_t)NEP*512*2);
  size_t o_ef   = alloc((size_t)NEP*64*4);
  // head scratch aliases dead M/H1/H2 region (after edge MLP completes)
  size_t o_gathf  = o_m;                        // <= 25.7MB in 32MB
  size_t o_gathbf = o_h1;                       // <= 12.9MB
  size_t o_y1f    = o_h1 + ((size_t)16 << 20);  // <= 51.3MB
  size_t o_y1bf   = o_h1 + ((size_t)68 << 20);  // <= 25.7MB
  size_t o_y2f    = o_h1 + ((size_t)94 << 20);  // <= 51.3MB
  size_t o_y2n    = o_h1 + ((size_t)146 << 20); // <= 51.3MB (fits in H1+H2 region)

  float* ssum = (float*)(ws + o_ssum);
  float* cnt  = (float*)(ws + o_cnt);
  unsigned short* xbf = (unsigned short*)(ws + o_xbf);
  unsigned short* Mb  = (unsigned short*)(ws + o_m);
  unsigned short* H1  = (unsigned short*)(ws + o_h1);
  unsigned short* H2  = (unsigned short*)(ws + o_h2);
  float* EF = (float*)(ws + o_ef);
  float* meanb = (float*)(ws + o_mean);
  float* rstdb = (float*)(ws + o_rstd);
  float* part  = (float*)(ws + o_part);

  auto gemm_lds = [&](const unsigned short* A, const unsigned short* Bt, const float* bias,
                      float* Cf, unsigned short* Cbf, int Mp, int N, int K, int act) {
    int blocks = (Mp >> 6) * (N >> 8);
    k_gemm_lds<<<blocks, 256, 0, stream>>>(A, Bt, bias, Cf, Cbf, Mp, N, K, act);
  };
  auto gemm_sm = [&](const unsigned short* A, const unsigned short* Bt, const float* bias,
                     float* Cf, unsigned short* Cbf, int Mp, int N, int K, int act) {
    int waves = (Mp >> 5) * (N >> 6);
    k_gemm_bf16<<<(waves + 7) / 8, 256, 0, stream>>>(A, Bt, bias, Cf, Cbf, Mp, N, K, act);
  };
  auto convw = [&](const float* W, size_t off, int K, int N, int Kpad) {
    int tot = N * Kpad;
    k_conv_weight<<<(tot + 255) / 256, 256, 0, stream>>>(W, (unsigned short*)(ws + off), K, N, Kpad);
  };

  // ---- 1. feature extraction + segment mean ----
  hipMemsetAsync(ws + o_ssum, 0, (size_t)NNODES * 33 * 4, stream); // ssum + cnt contiguous
  k_conv_scatter<<<(4*256*256 + 255)/256, 256, 0, stream>>>(raw, sp_seg, cw, cb, ssum, cnt);
  k_node_feats<<<(NNODES*64 + 255)/256, 256, 0, stream>>>(ssum, cnt, sp_feat, xbf);

  // ---- 2. weight transposition to bf16 ----
  convw(g1w, o_wg1, 132, 512, 160);
  convw(g2w, o_wg2, 512, 512, 512);
  convw(g3w, o_wg3, 512, 64, 512);
  for (int i = 0; i < 3; i++) { convw(l1w[i], o_wl1[i], dim_d[i], 512, dim_d[i]);
                                convw(l2w[i], o_wl2[i], 512, 512, 512); }

  // ---- 3. edge message MLP (WMMA GEMMs) ----
  k_build_m<<<(NEP + 255)/256, 256, 0, stream>>>(xbf, edge_idx, actions, edge_ft, Mb);
  gemm_lds(Mb, (unsigned short*)(ws + o_wg1), g1b, nullptr, H1, NEP, 512, 160, 1);
  gemm_lds(H1, (unsigned short*)(ws + o_wg2), g2b, nullptr, H2, NEP, 512, 512, 1);
  gemm_sm (H2, (unsigned short*)(ws + o_wg3), g3b, EF, nullptr, NEP, 64, 512, 0);

  // ---- 4. side loss ----
  int nb = (NE + 255) / 256;
  k_score<<<nb, 256, 0, stream>>>(EF, scw, scb, gt, part);
  k_loss_final<<<1, 32, 0, stream>>>(part, nb, out + 43750);

  // ---- 5. subgraph value heads ----
  int ssg_arr[3] = {4, 8, 16};
  int out_off[3] = {0, 25000, 37500};
  float* gathf = (float*)(ws + o_gathf);
  unsigned short* gathbf = (unsigned short*)(ws + o_gathbf);
  float* y1f = (float*)(ws + o_y1f);
  unsigned short* y1bf = (unsigned short*)(ws + o_y1bf);
  float* y2f = (float*)(ws + o_y2f);
  float* y2n = (float*)(ws + o_y2n);

  for (int i = 0; i < 3; i++) {
    int ssg = ssg_arr[i], nsub = NE / ssg, d = 64 * ssg;
    int npad = (nsub + 63) & ~63;            // pad rows to 64 for guard-free GEMM
    int tot = npad * d;
    k_gather<<<(tot + 255)/256, 256, 0, stream>>>(EF, subg[i], gathf, nsub, npad, ssg);
    k_bn_stats<<<d, 256, 0, stream>>>(gathf, nsub, d, meanb, rstdb);
    k_bn_apply<<<(tot + 255)/256, 256, 0, stream>>>(gathf, meanb, rstdb, bn0g[i], bn0b[i],
                                                    gathbf, nullptr, npad, d);
    gemm_lds(gathbf, (unsigned short*)(ws + o_wl1[i]), l1b[i], y1f, nullptr, npad, 512, d, 0);
    k_bn_stats<<<512, 256, 0, stream>>>(y1f, nsub, 512, meanb, rstdb);
    k_bn_apply<<<(npad*512 + 255)/256, 256, 0, stream>>>(y1f, meanb, rstdb, bn1g[i], bn1b[i],
                                                         y1bf, nullptr, npad, 512);
    gemm_lds(y1bf, (unsigned short*)(ws + o_wl2[i]), l2b[i], y2f, nullptr, npad, 512, 512, 0);
    k_bn_stats<<<512, 256, 0, stream>>>(y2f, nsub, 512, meanb, rstdb);
    k_bn_apply<<<(npad*512 + 255)/256, 256, 0, stream>>>(y2f, meanb, rstdb, bn2g[i], bn2b[i],
                                                         nullptr, y2n, npad, 512);
    k_matvec512<<<(nsub + 7)/8, 256, 0, stream>>>(y2n, l3w[i], l3b[i], out + out_off[i], nsub);
  }
}
